// LayerWiseAttention_8108898255619
// MI455X (gfx1250) — matrix operand order repaired and checked by reading, not compiled, not run on hardware
//
#include <hip/hip_runtime.h>

// Problem constants (fixed by the reference)
#define S_   1024
#define B_   2
#define D_   1024
#define H_   16
#define HD_  64
#define LP1  4
#define BH_  (B_*H_)   // 32
#define M_   (S_*B_)   // 2048

typedef __attribute__((ext_vector_type(16))) __bf16         v16bf;
typedef __attribute__((ext_vector_type(8)))  float          v8f;
typedef __attribute__((ext_vector_type(4)))  float          f32x4;
typedef __attribute__((ext_vector_type(8)))  unsigned short u16x8;
typedef __attribute__((ext_vector_type(16))) unsigned short u16x16;

static __device__ __forceinline__ unsigned short f2bf(float f) {
  unsigned int u = __builtin_bit_cast(unsigned int, f);
  u += 0x7FFFu + ((u >> 16) & 1u);          // round-to-nearest-even
  return (unsigned short)(u >> 16);
}
static __device__ __forceinline__ __bf16 bfbits(unsigned short s) {
  return __builtin_bit_cast(__bf16, s);
}

// Pack 16 fp32 -> v16bf; prefer hardware v_cvt_pk_bf16_f32 when available.
static __device__ __forceinline__ v16bf pack16(const float* f) {
  v16bf a;
#if __has_builtin(__builtin_amdgcn_cvt_pk_bf16_f32)
  typedef __attribute__((ext_vector_type(2))) __bf16 bf16x2;
  #pragma unroll
  for (int i = 0; i < 8; ++i) {
    bf16x2 p = __builtin_amdgcn_cvt_pk_bf16_f32(f[2*i], f[2*i+1]);
    a[2*i] = p[0]; a[2*i+1] = p[1];
  }
#else
  #pragma unroll
  for (int i = 0; i < 16; ++i) a[i] = bfbits(f2bf(f[i]));
#endif
  return a;
}

// XOR lane-shuffle via ds_swizzle (group-of-32: and=0x1F, or=0, xor=MASK).
// Masks 1,2,4,8 reduce within each 16-lane half independently (== row reduce
// for the WMMA C/D layout: col = lane%16, row = vgpr + 8*(lane/16)).
template<int MASK>
static __device__ __forceinline__ float swz(float v) {
  return __builtin_bit_cast(float,
      __builtin_amdgcn_ds_swizzle(__builtin_bit_cast(int, v), 0x1F | (MASK << 10)));
}
static __device__ __forceinline__ float rowmax16(float v) {
  v = fmaxf(v, swz<1>(v)); v = fmaxf(v, swz<2>(v));
  v = fmaxf(v, swz<4>(v)); v = fmaxf(v, swz<8>(v));
  return v;
}
static __device__ __forceinline__ float rowsum16(float v) {
  v += swz<1>(v); v += swz<2>(v); v += swz<4>(v); v += swz<8>(v);
  return v;
}

// ---------------------------------------------------------------------------
// WMMA GEMM: C[M_,D_] = A[M_,D_] @ W[D_,D_]^T + bias
// One wave computes a 16x64 strip (4 N-tiles): A operand loaded once per
// k-chunk, reused by 4 back-to-back WMMAs.
// AMODE: 0 = A fp32 (convert), 1 = A bf16 bits
// OMODE: 0 = bf16 [BH][S][hd] (Q/K), 1 = bf16 [BH][hd][S] (V^T),
//        2 = fp32 row-major [M_][D_] (final projection)
// ---------------------------------------------------------------------------
template<int AMODE, int OMODE>
__global__ __launch_bounds__(256) void gemm_wmma(const void* __restrict__ Av,
                                                 const float* __restrict__ W,
                                                 const float* __restrict__ bias,
                                                 void* __restrict__ Out) {
  const int wid  = threadIdx.x >> 5;
  const int lane = threadIdx.x & 31;
  const int lrow = lane & 15;
  const int lhi  = lane >> 4;
  const int t    = blockIdx.x * 8 + wid;    // wave job id
  const int NG   = D_ / 64;                 // 16 strip groups along N
  const int mt   = t / NG;                  // 0..127
  const int ng   = t % NG;                  // 0..15
  if (mt >= M_ / 16) return;

  const int arow = mt * 16 + lrow;

  const float*          Af = (const float*)Av;
  const unsigned short* Ab = (const unsigned short*)Av;

  v8f acc[4] = {};
  #pragma unroll 2
  for (int kb = 0; kb < D_; kb += 32) {
    // ---- A operand (16x32 bf16): lane K = kb+8*lhi+0..7, kb+16+8*lhi+0..7
    v16bf a;
    if (AMODE == 0) {
      const float* ar = Af + (size_t)arow * D_ + kb;
      f32x4 a0 = *(const f32x4*)(ar + 8 * lhi);
      f32x4 a1 = *(const f32x4*)(ar + 8 * lhi + 4);
      f32x4 a2 = *(const f32x4*)(ar + 16 + 8 * lhi);
      f32x4 a3 = *(const f32x4*)(ar + 16 + 8 * lhi + 4);
      float fa[16];
      #pragma unroll
      for (int i = 0; i < 4; ++i) {
        fa[i] = a0[i]; fa[i+4] = a1[i]; fa[i+8] = a2[i]; fa[i+12] = a3[i];
      }
      a = pack16(fa);
    } else {
      const unsigned short* ar = Ab + (size_t)arow * D_ + kb;
      u16x8 a0 = *(const u16x8*)(ar + 8 * lhi);
      u16x8 a1 = *(const u16x8*)(ar + 16 + 8 * lhi);
      u16x16 au = __builtin_shufflevector(a0, a1, 0,1,2,3,4,5,6,7,8,9,10,11,12,13,14,15);
      a = __builtin_bit_cast(v16bf, au);
    }
    // ---- 4 B operands (32x16 = W^T chunks), 4 WMMAs reusing A
    #pragma unroll
    for (int i = 0; i < 4; ++i) {
      const int    ncol = (ng * 4 + i) * 16 + lrow;
      const float* br   = W + (size_t)ncol * D_ + kb + 8 * lhi;
      f32x4 b0 = *(const f32x4*)(br);
      f32x4 b1 = *(const f32x4*)(br + 4);
      f32x4 b2 = *(const f32x4*)(br + 16);
      f32x4 b3 = *(const f32x4*)(br + 20);
      float fb[16];
      #pragma unroll
      for (int k = 0; k < 4; ++k) {
        fb[k] = b0[k]; fb[k+4] = b1[k]; fb[k+8] = b2[k]; fb[k+12] = b3[k];
      }
      v16bf b = pack16(fb);
      acc[i] = __builtin_amdgcn_wmma_f32_16x16x32_bf16(false, a, false, b, (short)0,
                                                       acc[i], false, false);
    }
  }

  // ---- epilogue
  #pragma unroll
  for (int i = 0; i < 4; ++i) {
    const int   ncol = (ng * 4 + i) * 16 + lrow;
    const float bvs  = bias[ncol];
    #pragma unroll
    for (int j = 0; j < 8; ++j) {
      const int   m   = mt * 16 + j + 8 * lhi;   // C row (D layout)
      const float val = acc[i][j] + bvs;
      if (OMODE == 2) {
        ((float*)Out)[(size_t)m * D_ + ncol] = val;
      } else {
        const int s  = m >> 1;                   // m = s*B + b, B=2
        const int bb = m & 1;
        const int h  = ncol >> 6;                // ncol = h*64 + d
        const int d  = ncol & 63;
        const int bh = bb * H_ + h;
        unsigned short* o = (unsigned short*)Out;
        if (OMODE == 0) o[((size_t)(bh * S_ + s)) * HD_ + d] = f2bf(val);
        else            o[((size_t)(bh * HD_ + d)) * S_ + s] = f2bf(val);
      }
    }
  }
}

// ---------------------------------------------------------------------------
// Fused layer-wise flash attention (one wave = one (bh, 16-query tile)).
// 4 layers x 64 key tiles (pairs -> K=32 PV WMMA); online softmax with
// ds_swizzle XOR row reductions; P transposed D->A layout via per-wave LDS.
// ---------------------------------------------------------------------------
__global__ __launch_bounds__(128) void attn_wmma(const unsigned short* __restrict__ Q,
                                                 const unsigned short* __restrict__ Kb,
                                                 const unsigned short* __restrict__ Vt,
                                                 const float* __restrict__ LWp,
                                                 unsigned short* __restrict__ Comb) {
  __shared__ __attribute__((aligned(16))) unsigned short lds[4][16 * 32];
  const int wid  = threadIdx.x >> 5;
  const int lane = threadIdx.x & 31;
  const int lrow = lane & 15;
  const int lhi  = lane >> 4;
  const int g    = blockIdx.x * 4 + wid;
  const int bh   = g >> 6;                 // 64 query tiles per bh
  const int qt   = g & 63;

  // softmax over first Lp1=4 layer weights (uniform across lanes)
  float w0 = LWp[0], w1 = LWp[1], w2 = LWp[2], w3 = LWp[3];
  float wm = fmaxf(fmaxf(w0, w1), fmaxf(w2, w3));
  float e0 = __expf(w0 - wm), e1 = __expf(w1 - wm);
  float e2 = __expf(w2 - wm), e3 = __expf(w3 - wm);
  float es = e0 + e1 + e2 + e3;
  float lwn[4] = {e0 / es, e1 / es, e2 / es, e3 / es};

  // Q A-operands for hd=64: two 16x32 chunks
  const unsigned short* qrow = Q + ((size_t)(bh * S_ + qt * 16 + lrow)) * HD_;
  v16bf qa[2];
  #pragma unroll
  for (int kc = 0; kc < 2; ++kc) {
    u16x8 a0 = *(const u16x8*)(qrow + kc * 32 + 16 * lhi);
    u16x8 a1 = *(const u16x8*)(qrow + kc * 32 + 16 * lhi + 8);
    u16x16 au = __builtin_shufflevector(a0, a1, 0,1,2,3,4,5,6,7,8,9,10,11,12,13,14,15);
    qa[kc] = __builtin_bit_cast(v16bf, au);
  }

  v8f tot[4] = {};                          // fp32 total over layers, 16x64
  unsigned short* slab = lds[wid];

  #pragma unroll
  for (int l = 0; l < LP1; ++l) {
    const unsigned short* kl = Kb + ((size_t)(l * BH_ + bh)) * S_ * HD_;
    const unsigned short* vl = Vt + ((size_t)(l * BH_ + bh)) * HD_ * S_;

    v8f   accL[4] = {};
    float mrow[8], lsum[8];
    #pragma unroll
    for (int j = 0; j < 8; ++j) { mrow[j] = -3.0e38f; lsum[j] = 0.f; }

    for (int kt = 0; kt < 64; kt += 2) {
      // prefetch next pair of key rows (streamed exactly once)
      if (kt + 2 < 64) {
        __builtin_prefetch(kl + ((size_t)((kt + 2) * 16 + lrow)) * HD_ + 16 * lhi, 0, 1);
        __builtin_prefetch(kl + ((size_t)((kt + 3) * 16 + lrow)) * HD_ + 16 * lhi, 0, 1);
      }
      // ---- scores for key tiles kt, kt+1 : 2 WMMAs each over hd chunks
      v8f s0 = {}, s1 = {};
      const unsigned short* kr0 = kl + ((size_t)((kt + 0) * 16 + lrow)) * HD_ + 16 * lhi;
      const unsigned short* kr1 = kl + ((size_t)((kt + 1) * 16 + lrow)) * HD_ + 16 * lhi;
      #pragma unroll
      for (int kc = 0; kc < 2; ++kc) {
        v16bf b0 = __builtin_bit_cast(v16bf, *(const u16x16*)(kr0 + kc * 32));
        v16bf b1 = __builtin_bit_cast(v16bf, *(const u16x16*)(kr1 + kc * 32));
        s0 = __builtin_amdgcn_wmma_f32_16x16x32_bf16(false, qa[kc], false, b0, (short)0,
                                                     s0, false, false);
        s1 = __builtin_amdgcn_wmma_f32_16x16x32_bf16(false, qa[kc], false, b1, (short)0,
                                                     s1, false, false);
      }
      #pragma unroll
      for (int j = 0; j < 8; ++j) { s0[j] *= 0.125f; s1[j] *= 0.125f; }  // hd^-0.5

      // ---- online softmax (ds_swizzle XOR reductions over 16-lane halves)
      float corr[8];
      #pragma unroll
      for (int j = 0; j < 8; ++j) {
        float v  = rowmax16(fmaxf(s0[j], s1[j]));
        float mn = fmaxf(mrow[j], v);
        corr[j]  = __expf(mrow[j] - mn);
        mrow[j]  = mn;
        s0[j]    = __expf(s0[j] - mn);
        s1[j]    = __expf(s1[j] - mn);
        lsum[j]  = lsum[j] * corr[j] + rowsum16(s0[j] + s1[j]);
      }
      #pragma unroll
      for (int c = 0; c < 4; ++c)
        #pragma unroll
        for (int j = 0; j < 8; ++j) accL[c][j] *= corr[j];

      // ---- transpose P: D layout -> row-major 16x32 bf16 in LDS
      #pragma unroll
      for (int j = 0; j < 8; ++j) {
        const int row = j + 8 * lhi;
        slab[row * 32 + lrow]      = f2bf(s0[j]);
        slab[row * 32 + 16 + lrow] = f2bf(s1[j]);
      }
      asm volatile("s_wait_dscnt 0x0" ::: "memory");
      u16x8 pa0 = *(const u16x8*)(slab + lrow * 32 + 16 * lhi);
      u16x8 pa1 = *(const u16x8*)(slab + lrow * 32 + 16 * lhi + 8);
      u16x16 pu = __builtin_shufflevector(pa0, pa1, 0,1,2,3,4,5,6,7,8,9,10,11,12,13,14,15);
      v16bf pA  = __builtin_bit_cast(v16bf, pu);

      // ---- PV: 4 channel tiles, B from V^T contiguous per lane
      #pragma unroll
      for (int c = 0; c < 4; ++c) {
        const unsigned short* vr = vl + ((size_t)(c * 16 + lrow)) * S_ + kt * 16 + 16 * lhi;
        v16bf bv = __builtin_bit_cast(v16bf, *(const u16x16*)vr);
        accL[c] = __builtin_amdgcn_wmma_f32_16x16x32_bf16(false, pA, false, bv, (short)0,
                                                          accL[c], false, false);
      }
    }

    // fold layer into total: lw_l * (accL / rowsum)
    float inv[8];
    #pragma unroll
    for (int j = 0; j < 8; ++j) inv[j] = lwn[l] / lsum[j];
    #pragma unroll
    for (int c = 0; c < 4; ++c)
      #pragma unroll
      for (int j = 0; j < 8; ++j) tot[c][j] += accL[c][j] * inv[j];
  }

  // ---- store combined as bf16 [S*B, D] row-major for the final GEMM
  const int bb = bh >> 4, h = bh & 15;
  #pragma unroll
  for (int c = 0; c < 4; ++c) {
    const int n = h * 64 + c * 16 + lrow;
    #pragma unroll
    for (int j = 0; j < 8; ++j) {
      const int s = qt * 16 + j + 8 * lhi;
      const int m = s * B_ + bb;
      Comb[(size_t)m * D_ + n] = f2bf(tot[c][j]);
    }
  }
}

// ---------------------------------------------------------------------------
extern "C" void kernel_launch(void* const* d_in, const int* in_sizes, int n_in,
                              void* d_out, int out_size, void* d_ws, size_t ws_size,
                              hipStream_t stream) {
  (void)in_sizes; (void)n_in; (void)out_size; (void)ws_size;
  const float* x  = (const float*)d_in[0];
  const float* lo = (const float*)d_in[1];
  const float* Wq = (const float*)d_in[2];  const float* bq = (const float*)d_in[3];
  const float* Wk = (const float*)d_in[4];  const float* bk = (const float*)d_in[5];
  const float* Wv = (const float*)d_in[6];  const float* bv = (const float*)d_in[7];
  const float* Wo = (const float*)d_in[8];  const float* bo = (const float*)d_in[9];
  const float* lw = (const float*)d_in[10];
  float* out = (float*)d_out;

  // workspace: bf16 buffers (each 2M elems = 4 MB): Q | K[4] | Vt[4] | Comb = 40 MB
  const size_t SEG = (size_t)BH_ * S_ * HD_;          // 2,097,152 elems
  unsigned short* Qb   = (unsigned short*)d_ws;
  unsigned short* Kbuf = Qb   + SEG;
  unsigned short* Vtb  = Kbuf + (size_t)LP1 * SEG;
  unsigned short* Comb = Vtb  + (size_t)LP1 * SEG;

  const dim3 gblk(256);
  const dim3 ggrid((M_ / 16) * (D_ / 64) / 8);        // 256 blocks, 8 waves each

  gemm_wmma<0, 0><<<ggrid, gblk, 0, stream>>>(x, Wq, bq, Qb);
  for (int l = 0; l < LP1; ++l) {
    const float* act = (l == 0) ? x : (lo + (size_t)(l - 1) * M_ * D_);
    gemm_wmma<0, 0><<<ggrid, gblk, 0, stream>>>(act, Wk, bk, Kbuf + (size_t)l * SEG);
    gemm_wmma<0, 1><<<ggrid, gblk, 0, stream>>>(act, Wv, bv, Vtb  + (size_t)l * SEG);
  }
  attn_wmma<<<dim3(BH_ * 64 / 4), dim3(128), 0, stream>>>(Qb, Kbuf, Vtb, lw, Comb);
  gemm_wmma<1, 2><<<ggrid, gblk, 0, stream>>>(Comb, Wo, bo, out);
}
